// PCFG_Encoder_44925357916233
// MI455X (gfx1250) — compile-verified, matrix operand order checked by reading, not run on hardware
//
#include <hip/hip_runtime.h>

// ---------------- problem constants ----------------
#define Bz 4
#define Tz 1024
#define Cz 1024
#define Lz 4
#define Hz 8
#define Dz 128          // head dim
#define Fz 4096         // ffn hidden
#define LN_EPS 1e-5f

// ---------------- WMMA types ----------------
typedef __bf16 v16bf __attribute__((ext_vector_type(16)));
typedef __bf16 v2bf  __attribute__((ext_vector_type(2)));
typedef float  v8f   __attribute__((ext_vector_type(8)));
typedef int    v8i   __attribute__((ext_vector_type(8)));

// fp32 -> packed bf16 pair (RNE); plain casts let the backend pick
// native cvt instructions (v_cvt_pk_bf16_f32) on gfx1250.
__device__ __forceinline__ unsigned int pack2bf(float a, float b) {
    v2bf r;
    r[0] = (__bf16)a;
    r[1] = (__bf16)b;
    return __builtin_bit_cast(unsigned int, r);
}

// ---------------- generic batched GEMM: Out = [res +] A*B [+ bias] [relu] ----
// A: fp32 row-major [M,K], stride lda.  B: fp32, (k,n) at B[k*ldb+n] or,
// if transB, at B[n*ldb+k].  Batched over grid.z = batchH*batchB with
// independent (h,b) pointer strides for A, B, Out/res.
struct GemmP {
    const float* A;  long lda; long sAh; long sAb;
    const float* Bm; long ldb; long sBh; long sBb; int transB;
    float* Out;      long ldc; long sOh; long sOb;
    const float* bias; const float* res; int relu;
    int M, N, K, batchB;
};

#define BM 128
#define BN 128
#define BK 32
#define LDP (BK + 2)   // even pad -> dword-aligned pair reads

__global__ __launch_bounds__(256)
void wmma_gemm(GemmP p) {
    // double-buffered operand tiles (bf16)
    __shared__ unsigned short As[2][BM][LDP];   // [buf][row][k]
    __shared__ unsigned short Bs[2][BN][LDP];   // [buf][col][k]  (B^T tile)

    const int z  = blockIdx.z;
    const int hb = z / p.batchB;
    const int bb = z % p.batchB;
    const float* A  = p.A  + (size_t)hb * p.sAh + (size_t)bb * p.sAb;
    const float* Bg = p.Bm + (size_t)hb * p.sBh + (size_t)bb * p.sBb;
    float*       Og = p.Out + (size_t)hb * p.sOh + (size_t)bb * p.sOb;
    const float* Rg = p.res ? p.res + (size_t)hb * p.sOh + (size_t)bb * p.sOb : nullptr;

    const int n0   = blockIdx.x * BN;
    const int m0   = blockIdx.y * BM;
    const int tid  = threadIdx.x;
    const int lane = tid & 31;
    const int wave = tid >> 5;
    const int waveM = wave & 3;   // 4 groups of 32 rows
    const int waveN = wave >> 2;  // 2 groups of 64 cols
    const int half = lane >> 4;
    const int l16  = lane & 15;

    // staging index precompute
    const int arow = tid >> 3;            // A / transB: row(col), 4-wide k chunk
    const int akc  = (tid & 7) << 2;
    const int bn   = tid & 127;           // B (no-trans): one column per thread
    const int bkc0 = (tid >> 7) << 2;     // 0 or 4

    const v8f vzero = {0.f,0.f,0.f,0.f,0.f,0.f,0.f,0.f};
    v8f acc[2][4];
#pragma unroll
    for (int i = 0; i < 2; ++i)
#pragma unroll
        for (int j = 0; j < 4; ++j) acc[i][j] = vzero;

    float4 ra[4], rb[4];

    auto loadA = [&](int k0) {
#pragma unroll
        for (int it = 0; it < 4; ++it) {
            const int row = arow + it * 32;
            ra[it] = *(const float4*)(A + (size_t)(m0 + row) * p.lda + (k0 + akc));
        }
    };
    auto storeA = [&](int buf) {
#pragma unroll
        for (int it = 0; it < 4; ++it) {
            const int row = arow + it * 32;
            *(unsigned int*)&As[buf][row][akc]     = pack2bf(ra[it].x, ra[it].y);
            *(unsigned int*)&As[buf][row][akc + 2] = pack2bf(ra[it].z, ra[it].w);
        }
    };
    auto loadB = [&](int k0) {
        if (p.transB) {
#pragma unroll
            for (int it = 0; it < 4; ++it) {
                const int n = arow + it * 32;
                rb[it] = *(const float4*)(Bg + (size_t)(n0 + n) * p.ldb + (k0 + akc));
            }
        } else {
            // column-owner staging: lane-consecutive n -> coalesced b32 loads
#pragma unroll
            for (int it = 0; it < 4; ++it) {
                const int kc = bkc0 + it * 8;
                const float* base = Bg + (size_t)(k0 + kc) * p.ldb + (n0 + bn);
                rb[it].x = base[0];
                rb[it].y = base[(size_t)p.ldb];
                rb[it].z = base[(size_t)p.ldb * 2];
                rb[it].w = base[(size_t)p.ldb * 3];
            }
        }
    };
    auto storeB = [&](int buf) {
        if (p.transB) {
#pragma unroll
            for (int it = 0; it < 4; ++it) {
                const int n = arow + it * 32;
                *(unsigned int*)&Bs[buf][n][akc]     = pack2bf(rb[it].x, rb[it].y);
                *(unsigned int*)&Bs[buf][n][akc + 2] = pack2bf(rb[it].z, rb[it].w);
            }
        } else {
#pragma unroll
            for (int it = 0; it < 4; ++it) {
                const int kc = bkc0 + it * 8;
                *(unsigned int*)&Bs[buf][bn][kc]     = pack2bf(rb[it].x, rb[it].y);
                *(unsigned int*)&Bs[buf][bn][kc + 2] = pack2bf(rb[it].z, rb[it].w);
            }
        }
    };

    const int ktiles = p.K / BK;

    // prologue: stage tile 0 into buffer 0
    loadA(0);
    loadB(0);
    storeA(0);
    storeB(0);
    __syncthreads();

    for (int kt = 0; kt < ktiles; ++kt) {
        const int cur = kt & 1;
        const int nxt = cur ^ 1;
        const bool more = (kt + 1) < ktiles;

        // issue next tile's global loads early (in flight during WMMA)
        if (more) { loadA((kt + 1) * BK); loadB((kt + 1) * BK); }

        // ---- build fragments (ISA 7.12.2 16-bit A/B layout) ----
        v16bf afrag[2], bfrag[4];
#pragma unroll
        for (int tm = 0; tm < 2; ++tm) {
            const unsigned short* src = &As[cur][waveM * 32 + tm * 16 + l16][0];
            v8i iv;
#pragma unroll
            for (int pp = 0; pp < 8; ++pp) {
                const int kk = (pp < 4 ? 2 * pp : 2 * pp + 8) + 8 * half;
                iv[pp] = *(const int*)(src + kk);
            }
            afrag[tm] = __builtin_bit_cast(v16bf, iv);
        }
#pragma unroll
        for (int tn = 0; tn < 4; ++tn) {
            const unsigned short* src = &Bs[cur][waveN * 64 + tn * 16 + l16][0];
            v8i iv;
#pragma unroll
            for (int pp = 0; pp < 8; ++pp) {
                const int kk = (pp < 4 ? 2 * pp : 2 * pp + 8) + 8 * half;
                iv[pp] = *(const int*)(src + kk);
            }
            bfrag[tn] = __builtin_bit_cast(v16bf, iv);
        }
        // ---- 8 WMMAs per wave per K-step ----
#pragma unroll
        for (int tm = 0; tm < 2; ++tm)
#pragma unroll
            for (int tn = 0; tn < 4; ++tn)
                acc[tm][tn] = __builtin_amdgcn_wmma_f32_16x16x32_bf16(
                    false, afrag[tm], false, bfrag[tn],
                    (short)0, acc[tm][tn], false, false);

        // convert+store the prefetched tile into the other buffer
        if (more) { storeA(nxt); storeB(nxt); }
        __syncthreads();
    }

    // ---- epilogue: bias / relu / residual, fp32 store ----
#pragma unroll
    for (int tm = 0; tm < 2; ++tm) {
#pragma unroll
        for (int tn = 0; tn < 4; ++tn) {
            const int nG = n0 + waveN * 64 + tn * 16 + l16;
            const float bv = p.bias ? p.bias[nG] : 0.0f;
#pragma unroll
            for (int e = 0; e < 8; ++e) {
                const int rG = m0 + waveM * 32 + tm * 16 + e + 8 * half;
                float val = acc[tm][tn][e] + bv;
                if (p.relu) val = fmaxf(val, 0.0f);
                const size_t off = (size_t)rG * p.ldc + nG;
                if (Rg) val += Rg[off];
                Og[off] = val;
            }
        }
    }
}

// ---------------- wave32 reductions ----------------
__device__ __forceinline__ float wred_add(float v) {
#pragma unroll
    for (int off = 16; off > 0; off >>= 1) v += __shfl_xor(v, off, 32);
    return v;
}
__device__ __forceinline__ float wred_max(float v) {
#pragma unroll
    for (int off = 16; off > 0; off >>= 1) v = fmaxf(v, __shfl_xor(v, off, 32));
    return v;
}

// ---------------- elementwise / row kernels (fp32) ----------------
__global__ __launch_bounds__(256)
void addpos_k(const float* __restrict__ x, const float* __restrict__ pos,
              float* __restrict__ out) {
    const size_t i = (size_t)blockIdx.x * 256 + threadIdx.x;
    const size_t tc = i % ((size_t)Tz * Cz);
    out[i] = x[i] + pos[tc];
}

__global__ __launch_bounds__(256)
void layernorm_k(const float* __restrict__ x, const float* __restrict__ s,
                 const float* __restrict__ b, float* __restrict__ out) {
    __shared__ float redS[8], redQ[8];
    const int row = blockIdx.x;
    const int lane = threadIdx.x & 31;
    const int wid  = threadIdx.x >> 5;
    const float* xr = x + (size_t)row * Cz;
    float v[4], lsum = 0.f, lsq = 0.f;
#pragma unroll
    for (int i = 0; i < 4; ++i) {
        v[i] = xr[threadIdx.x + i * 256];
        lsum += v[i]; lsq += v[i] * v[i];
    }
    const float ws = wred_add(lsum);
    const float wq = wred_add(lsq);
    if (lane == 0) { redS[wid] = ws; redQ[wid] = wq; }
    __syncthreads();
    float tS = 0.f, tQ = 0.f;
#pragma unroll
    for (int i = 0; i < 8; ++i) { tS += redS[i]; tQ += redQ[i]; }
    const float mean = tS * (1.0f / Cz);
    const float var  = tQ * (1.0f / Cz) - mean * mean;
    const float inv  = rsqrtf(var + LN_EPS);
    float* orow = out + (size_t)row * Cz;
#pragma unroll
    for (int i = 0; i < 4; ++i) {
        const int c = threadIdx.x + i * 256;
        orow[c] = (v[i] - mean) * inv * s[c] + b[c];
    }
}

__global__ __launch_bounds__(256)
void softmax_k(float* __restrict__ att, float scale) {
    __shared__ float red[8];
    const int lane = threadIdx.x & 31;
    const int wid  = threadIdx.x >> 5;
    float* r = att + (size_t)blockIdx.x * Tz;
    float v[4];
    float lmax = -3.4e38f;
#pragma unroll
    for (int i = 0; i < 4; ++i) {
        v[i] = r[threadIdx.x + i * 256] * scale;
        lmax = fmaxf(lmax, v[i]);
    }
    const float wm = wred_max(lmax);
    if (lane == 0) red[wid] = wm;
    __syncthreads();
    float m = -3.4e38f;
#pragma unroll
    for (int i = 0; i < 8; ++i) m = fmaxf(m, red[i]);
    __syncthreads();
    float lsum = 0.f;
#pragma unroll
    for (int i = 0; i < 4; ++i) { v[i] = __expf(v[i] - m); lsum += v[i]; }
    const float wsum = wred_add(lsum);
    if (lane == 0) red[wid] = wsum;
    __syncthreads();
    float tot = 0.f;
#pragma unroll
    for (int i = 0; i < 8; ++i) tot += red[i];
    const float inv = 1.0f / tot;
#pragma unroll
    for (int i = 0; i < 4; ++i) r[threadIdx.x + i * 256] = v[i] * inv;
}

// ---------------- host orchestration ----------------
static void launch_gemm(hipStream_t stream,
                        const float* A, long lda, long sAh, long sAb,
                        const float* Bm, long ldb, long sBh, long sBb, int transB,
                        float* Out, long ldc, long sOh, long sOb,
                        const float* bias, const float* res, int relu,
                        int M, int N, int K, int batchH, int batchB) {
    GemmP p;
    p.A = A; p.lda = lda; p.sAh = sAh; p.sAb = sAb;
    p.Bm = Bm; p.ldb = ldb; p.sBh = sBh; p.sBb = sBb; p.transB = transB;
    p.Out = Out; p.ldc = ldc; p.sOh = sOh; p.sOb = sOb;
    p.bias = bias; p.res = res; p.relu = relu;
    p.M = M; p.N = N; p.K = K; p.batchB = batchB;
    dim3 grid(N / BN, M / BM, batchH * batchB);
    wmma_gemm<<<grid, dim3(256), 0, stream>>>(p);
}

extern "C" void kernel_launch(void* const* d_in, const int* in_sizes, int n_in,
                              void* d_out, int out_size, void* d_ws, size_t ws_size,
                              hipStream_t stream) {
    (void)in_sizes; (void)n_in; (void)out_size; (void)ws_size;
    const float* in_x = (const float*)d_in[0];
    const float* pos  = (const float*)d_in[1];
    const float* Wq   = (const float*)d_in[2];
    const float* Wk   = (const float*)d_in[3];
    const float* Wv   = (const float*)d_in[4];
    const float* Wo   = (const float*)d_in[5];
    const float* bo   = (const float*)d_in[6];
    const float* ln1s = (const float*)d_in[7];
    const float* ln1b = (const float*)d_in[8];
    const float* ln2s = (const float*)d_in[9];
    const float* ln2b = (const float*)d_in[10];
    const float* W1   = (const float*)d_in[11];
    const float* b1   = (const float*)d_in[12];
    const float* W2   = (const float*)d_in[13];
    const float* b2   = (const float*)d_in[14];
    const float* lnfs = (const float*)d_in[15];
    const float* lnfb = (const float*)d_in[16];

    const size_t BTC = (size_t)Bz * Tz * Cz;            // 4M
    const size_t attsL = (size_t)Hz * Bz * Tz * Tz;     // per-layer atts elems

    float* x    = (float*)d_ws;            // [B,T,C]
    float* xn   = x    + BTC;              // [B,T,C]
    float* obuf = xn   + BTC;              // [B,T,C]
    float* big  = obuf + BTC;              // 16M floats
    float* q    = big;                     // [B,T,C] (head-major cols)
    float* kbuf = big + BTC;
    float* vbuf = big + 2 * BTC;
    float* ff   = big;                     // [B*T, F] reuses q/k/v region

    float* outx = (float*)d_out;           // [B,T,C]
    float* atts = outx + BTC;              // [L,H,B,T,T]

    const float scale = 0.08838834764831845f;  // D^-0.5
    const long rowsBT = (long)Bz * Tz;

    addpos_k<<<(int)(BTC / 256), 256, 0, stream>>>(in_x, pos, x);

    for (int l = 0; l < Lz; ++l) {
        const float* Wq_l = Wq + (size_t)l * Hz * Cz * Dz;
        const float* Wk_l = Wk + (size_t)l * Hz * Cz * Dz;
        const float* Wv_l = Wv + (size_t)l * Hz * Cz * Dz;
        const float* Wo_l = Wo + (size_t)l * Cz * Cz;
        const float* W1_l = W1 + (size_t)l * Cz * Fz;
        const float* W2_l = W2 + (size_t)l * Fz * Cz;
        float* attL = atts + (size_t)l * attsL;

        // LN1
        layernorm_k<<<(int)rowsBT, 256, 0, stream>>>(x, ln1s + l * Cz, ln1b + l * Cz, xn);

        // Q, K, V  (batched over heads: B-op stride C*D, out col stride D)
        launch_gemm(stream, xn, Cz, 0, 0, Wq_l, Dz, (long)Cz * Dz, 0, 0,
                    q,    Cz, Dz, 0, nullptr, nullptr, 0, (int)rowsBT, Dz, Cz, Hz, 1);
        launch_gemm(stream, xn, Cz, 0, 0, Wk_l, Dz, (long)Cz * Dz, 0, 0,
                    kbuf, Cz, Dz, 0, nullptr, nullptr, 0, (int)rowsBT, Dz, Cz, Hz, 1);
        launch_gemm(stream, xn, Cz, 0, 0, Wv_l, Dz, (long)Cz * Dz, 0, 0,
                    vbuf, Cz, Dz, 0, nullptr, nullptr, 0, (int)rowsBT, Dz, Cz, Hz, 1);

        // logits = q @ k^T  -> atts[l] (layout [H,B,T,T])
        launch_gemm(stream, q, Cz, Dz, (long)Tz * Cz,
                    kbuf, Cz, Dz, (long)Tz * Cz, 1,
                    attL, Tz, (long)Bz * Tz * Tz, (long)Tz * Tz,
                    nullptr, nullptr, 0, Tz, Tz, Dz, Hz, Bz);

        // softmax rows (scale folded in)
        softmax_k<<<Hz * Bz * Tz, 256, 0, stream>>>(attL, scale);

        // o = att @ v
        launch_gemm(stream, attL, Tz, (long)Bz * Tz * Tz, (long)Tz * Tz,
                    vbuf, Cz, Dz, (long)Tz * Cz, 0,
                    obuf, Cz, Dz, (long)Tz * Cz,
                    nullptr, nullptr, 0, Tz, Dz, Tz, Hz, Bz);

        // x = x + o @ Wo + bo
        launch_gemm(stream, obuf, Cz, 0, 0, Wo_l, Cz, 0, 0, 0,
                    x, Cz, 0, 0, bo + l * Cz, x, 0, (int)rowsBT, Cz, Cz, 1, 1);

        // LN2
        layernorm_k<<<(int)rowsBT, 256, 0, stream>>>(x, ln2s + l * Cz, ln2b + l * Cz, xn);

        // ff = relu(xn @ W1 + b1)
        launch_gemm(stream, xn, Cz, 0, 0, W1_l, Fz, 0, 0, 0,
                    ff, Fz, 0, 0, b1 + l * Fz, nullptr, 1, (int)rowsBT, Fz, Cz, 1, 1);

        // x = x + ff @ W2 + b2
        launch_gemm(stream, ff, Fz, 0, 0, W2_l, Cz, 0, 0, 0,
                    x, Cz, 0, 0, b2 + l * Cz, x, 0, (int)rowsBT, Cz, Fz, 1, 1);
    }

    // final LN -> d_out x region
    layernorm_k<<<(int)rowsBT, 256, 0, stream>>>(x, lnfs, lnfb, outx);
}
